// GATDualPhase_88450556494575
// MI455X (gfx1250) — matrix operand order repaired and checked by reading, not compiled, not run on hardware
//
#include <hip/hip_runtime.h>
#include <hip/hip_bf16.h>
#include <float.h>

// ---------------------------------------------------------------------------
// Types for CDNA5 WMMA (wave32): v_wmma_f32_16x16x32_bf16
// ---------------------------------------------------------------------------
typedef __attribute__((ext_vector_type(16))) __bf16          v16bf;
typedef __attribute__((ext_vector_type(16))) unsigned short  v16us;
typedef __attribute__((ext_vector_type(8)))  float           v8f;
typedef __attribute__((ext_vector_type(8)))  unsigned int    v8u;
typedef __attribute__((ext_vector_type(4)))  int             i32x4;

__device__ __forceinline__ unsigned short bfb(float f) {
    unsigned u = __float_as_uint(f);
    unsigned r = (u + 0x7FFFu + ((u >> 16) & 1u)) >> 16;   // round-to-nearest-even
    return (unsigned short)r;
}

// A-fragment (16x32 bf16, MxK): lane L holds row M=L%16.
// lanes 0-15: K = kc+{0..7, 16..23}; lanes 16-31: K = kc+{8..15, 24..31}.
// koff = (lane<16) ? 0 : 8.
__device__ __forceinline__ v16bf load_a_f32(const float* rowp, int kc, int koff) {
    const float4 q0 = *(const float4*)(rowp + kc + koff);
    const float4 q1 = *(const float4*)(rowp + kc + koff + 4);
    const float4 q2 = *(const float4*)(rowp + kc + 16 + koff);
    const float4 q3 = *(const float4*)(rowp + kc + 16 + koff + 4);
    v16us t;
    t[0]  = bfb(q0.x); t[1]  = bfb(q0.y); t[2]  = bfb(q0.z); t[3]  = bfb(q0.w);
    t[4]  = bfb(q1.x); t[5]  = bfb(q1.y); t[6]  = bfb(q1.z); t[7]  = bfb(q1.w);
    t[8]  = bfb(q2.x); t[9]  = bfb(q2.y); t[10] = bfb(q2.z); t[11] = bfb(q2.w);
    t[12] = bfb(q3.x); t[13] = bfb(q3.y); t[14] = bfb(q3.z); t[15] = bfb(q3.w);
    return __builtin_bit_cast(v16bf, t);
}

// B-fragment (32x16 bf16, KxN) from LDS holding W transposed: Wt[n][k], k contiguous.
// lane L -> column N = L%16; lanes 0-15: K = kc+{0..7, 16..23}, lanes 16-31: K = kc+{8..15, 24..31}.
__device__ __forceinline__ v16bf load_b_lds(const unsigned short* Wt, int col, int kc, int lane) {
    const unsigned short* p = Wt + col * 128 + kc + ((lane < 16) ? 0 : 8);
    const uint4 q0 = *(const uint4*)p;
    const uint4 q1 = *(const uint4*)(p + 16);
    v8u q = {q0.x, q0.y, q0.z, q0.w, q1.x, q1.y, q1.z, q1.w};   // 2x ds_load_b128
    return __builtin_bit_cast(v16bf, q);
}

// Async stage of a pre-packed bf16 weight image (16B per lane per iteration),
// tracked by ASYNCcnt (CDNA5 GLOBAL_LOAD_ASYNC_TO_LDS_B128).
__device__ __forceinline__ void stage_weights_async(const unsigned short* __restrict__ src,
                                                    unsigned short* dst_lds,
                                                    int bytes, int tid, int nthreads) {
    for (int ofs = tid * 16; ofs < bytes; ofs += nthreads * 16) {
        __builtin_amdgcn_global_load_async_to_lds_b128(
            (__attribute__((address_space(1))) i32x4*)((const char*)src + ofs),
            (__attribute__((address_space(3))) i32x4*)((char*)dst_lds + ofs), 0, 0);
    }
    asm volatile("s_wait_asynccnt 0x0" ::: "memory");
    __syncthreads();
}

// Ordered-uint encoding for fp32 atomic max
__device__ __forceinline__ unsigned encf(float f) {
    unsigned u = __float_as_uint(f);
    return (u & 0x80000000u) ? ~u : (u | 0x80000000u);
}
__device__ __forceinline__ float decf(unsigned e) {
    unsigned u = (e & 0x80000000u) ? (e & 0x7FFFFFFFu) : ~e;
    return __uint_as_float(u);
}
#define ENC_NEG 0x00800000u   // encf(-FLT_MAX)

// ---------------------------------------------------------------------------
// Pre-pack: W[128,Ncols] f32 row-major -> Wt[n*128+k] bf16 (transposed image)
// ---------------------------------------------------------------------------
__global__ void pack_wt_kernel(const float* __restrict__ W, unsigned short* __restrict__ Wt,
                               int Ncols) {
    int i = blockIdx.x * blockDim.x + threadIdx.x;
    if (i >= 128 * Ncols) return;
    int k = i / Ncols, n = i - k * Ncols;
    Wt[n * 128 + k] = bfb(W[i]);
}

// ---------------------------------------------------------------------------
// Generic WMMA GEMM: Y[N,Ncols] = act(bf16(X[N,128]) @ W + b), W pre-packed bf16
// ---------------------------------------------------------------------------
__global__ void wmma_gemm_kernel(const float* __restrict__ X,
                                 const unsigned short* __restrict__ Wt_g,
                                 const float* __restrict__ bias, float* __restrict__ Y,
                                 int N, int Ncols, int do_relu) {
    __shared__ __align__(32) unsigned short Wt[128 * 128];
    const int tid = threadIdx.x;
    stage_weights_async(Wt_g, Wt, Ncols * 128 * 2, tid, blockDim.x);

    const int wave = tid >> 5, lane = tid & 31;
    const int tile = blockIdx.x * (blockDim.x >> 5) + wave;
    if (tile * 16 >= N) return;

    const int lcol = lane & 15;
    const float* xrow = X + (size_t)(tile * 16 + lcol) * 128;
    const int koff = (lane < 16) ? 0 : 8;

    v16bf a[4];
#pragma unroll
    for (int kb = 0; kb < 4; kb++) a[kb] = load_a_f32(xrow, kb * 32, koff);

    const int mofs = (lane < 16) ? 0 : 8;
    const int ntiles = Ncols >> 4;
    for (int ct = 0; ct < ntiles; ct++) {
        v8f acc = {};
#pragma unroll
        for (int kb = 0; kb < 4; kb++) {
            v16bf b = load_b_lds(Wt, ct * 16 + lcol, kb * 32, lane);
            acc = __builtin_amdgcn_wmma_f32_16x16x32_bf16(false, a[kb], false, b,
                                                          (short)0, acc, false, false);
        }
        const int col = ct * 16 + lcol;
        const float bv = bias[col];
#pragma unroll
        for (int r = 0; r < 8; r++) {
            float v = acc[r] + bv;
            if (do_relu) v = v > 0.f ? v : 0.f;
            Y[(size_t)(tile * 16 + mofs + r) * Ncols + col] = v;
        }
    }
}

// ---------------------------------------------------------------------------
// Per-conv edge pipeline
// ---------------------------------------------------------------------------
__global__ void conv_init_kernel(float* acc, unsigned* m_enc, float* den, int nd) {
    int i = blockIdx.x * blockDim.x + threadIdx.x;
    if (i < nd * 128) acc[i] = 0.f;
    if (i < nd * 2) { m_enc[i] = ENC_NEG; den[i] = 0.f; }
}

__global__ void logits_kernel(const float* __restrict__ xl, const float* __restrict__ xr,
                              const int* __restrict__ gl, const int* __restrict__ gs,
                              const float* __restrict__ att, float* __restrict__ logits,
                              unsigned* __restrict__ m_enc, int E_) {
    __shared__ float s_att[128];
    if (threadIdx.x < 128) s_att[threadIdx.x] = att[threadIdx.x];
    __syncthreads();
    int i = blockIdx.x * blockDim.x + threadIdx.x;
    if (i >= E_ * 2) return;
    const int e = i >> 1, hd = i & 1;
    const int s = gl[e], d = gs[e];
    const float4* pl = (const float4*)(xl + (size_t)s * 128 + hd * 64);
    const float4* pr = (const float4*)(xr + (size_t)d * 128 + hd * 64);
    float acc = 0.f;
#pragma unroll
    for (int c4 = 0; c4 < 16; c4++) {
        float4 a = pl[c4], b = pr[c4];
        float v0 = a.x + b.x; v0 = v0 > 0.f ? v0 : 0.2f * v0;
        float v1 = a.y + b.y; v1 = v1 > 0.f ? v1 : 0.2f * v1;
        float v2 = a.z + b.z; v2 = v2 > 0.f ? v2 : 0.2f * v2;
        float v3 = a.w + b.w; v3 = v3 > 0.f ? v3 : 0.2f * v3;
        const float* at = s_att + hd * 64 + c4 * 4;
        acc += v0 * at[0] + v1 * at[1] + v2 * at[2] + v3 * at[3];
    }
    logits[i] = acc;
    atomicMax(&m_enc[(size_t)d * 2 + hd], encf(acc));
}

__global__ void mfin_kernel(unsigned* m_enc, int nd) {
    int i = blockIdx.x * blockDim.x + threadIdx.x;
    if (i >= nd * 2) return;
    unsigned u = m_enc[i];
    float m = (u == ENC_NEG) ? 0.f : decf(u);   // empty segments -> 0 (ref isfinite fixup)
    ((float*)m_enc)[i] = m;                      // decode in place
}

__global__ void exden_kernel(const float* __restrict__ logits, const float* __restrict__ mfl,
                             const int* __restrict__ gs, float* __restrict__ exb,
                             float* __restrict__ den, int E_) {
    int i = blockIdx.x * blockDim.x + threadIdx.x;
    if (i >= E_ * 2) return;
    const int e = i >> 1, hd = i & 1;
    const int d = gs[e];
    float ex = expf(logits[i] - mfl[(size_t)d * 2 + hd]);
    exb[i] = ex;
    atomicAdd(&den[(size_t)d * 2 + hd], ex);
}

__global__ void scatter_kernel(const float* __restrict__ xl, const float* __restrict__ exb,
                               const float* __restrict__ den, const int* __restrict__ gl,
                               const int* __restrict__ gs, float* __restrict__ acc, int E_) {
    int i = blockIdx.x * blockDim.x + threadIdx.x;
    if (i >= E_ * 2) return;
    const int e = i >> 1, hd = i & 1;
    const int s = gl[e], d = gs[e];
    const float alpha = exb[i] / (den[(size_t)d * 2 + hd] + 1e-16f);
    const float4* ps = (const float4*)(xl + (size_t)s * 128 + hd * 64);
    float* pd = acc + (size_t)d * 128 + hd * 64;
#pragma unroll
    for (int c4 = 0; c4 < 16; c4++) {
        float4 v = ps[c4];
        atomicAdd(pd + c4 * 4 + 0, alpha * v.x);
        atomicAdd(pd + c4 * 4 + 1, alpha * v.y);
        atomicAdd(pd + c4 * 4 + 2, alpha * v.z);
        atomicAdd(pd + c4 * 4 + 3, alpha * v.w);
    }
}

__global__ void headmean_kernel(const float* __restrict__ acc, const float* __restrict__ bias,
                                float* __restrict__ comb, int colofs,
                                float* __restrict__ aux, int nd) {
    int i = blockIdx.x * blockDim.x + threadIdx.x;
    if (i >= nd * 64) return;
    const int n = i >> 6, c = i & 63;
    float v = 0.5f * (acc[(size_t)n * 128 + c] + acc[(size_t)n * 128 + 64 + c]) + bias[c];
    comb[(size_t)n * 128 + colofs + c] = v;
    aux[i] = v;
}

// ---------------------------------------------------------------------------
// Edge MLP: y[e] = relu(concat(uf[src],mf[dst]) @ Wp1 + bp1) @ Wp2 + bp2
// 16 edges per wave; h gathered straight into WMMA A fragments.
// ---------------------------------------------------------------------------
__global__ void edge_mlp_kernel(const float* __restrict__ uf, const float* __restrict__ mf,
                                const int* __restrict__ src, const int* __restrict__ dst,
                                const unsigned short* __restrict__ W1t_g,
                                const float* __restrict__ bp1,
                                const float* __restrict__ Wp2, const float* __restrict__ bp2,
                                float* __restrict__ y, int E_) {
    __shared__ __align__(32) unsigned short W1t[64 * 128];
    __shared__ float s_bp1[64], s_wp2[64];
    if (threadIdx.x < 64) { s_bp1[threadIdx.x] = bp1[threadIdx.x]; s_wp2[threadIdx.x] = Wp2[threadIdx.x]; }
    stage_weights_async(W1t_g, W1t, 64 * 128 * 2, threadIdx.x, blockDim.x);

    const int wave = threadIdx.x >> 5, lane = threadIdx.x & 31;
    const int tile = blockIdx.x * (blockDim.x >> 5) + wave;
    if (tile * 16 >= E_) return;

    const int lcol = lane & 15;
    const int e = tile * 16 + lcol;
    const float* hu = uf + (size_t)src[e] * 64;
    const float* hm = mf + (size_t)dst[e] * 64;
    const int koff = (lane < 16) ? 0 : 8;

    v16bf a[4];
    a[0] = load_a_f32(hu, 0, koff);    // h[0..31]   = uf
    a[1] = load_a_f32(hu, 32, koff);   // h[32..63]  = uf
    a[2] = load_a_f32(hm, 0, koff);    // h[64..95]  = mf
    a[3] = load_a_f32(hm, 32, koff);   // h[96..127] = mf

    float ysum[8];
#pragma unroll
    for (int r = 0; r < 8; r++) ysum[r] = 0.f;

    for (int ct = 0; ct < 4; ct++) {
        v8f acc = {};
#pragma unroll
        for (int kb = 0; kb < 4; kb++) {
            v16bf b = load_b_lds(W1t, ct * 16 + lcol, kb * 32, lane);
            acc = __builtin_amdgcn_wmma_f32_16x16x32_bf16(false, a[kb], false, b,
                                                          (short)0, acc, false, false);
        }
        const int col = ct * 16 + lcol;
        const float bb = s_bp1[col], w2 = s_wp2[col];
#pragma unroll
        for (int r = 0; r < 8; r++) {
            float v = acc[r] + bb;
            v = v > 0.f ? v : 0.f;
            ysum[r] += v * w2;
        }
    }
    // reduce over the 16 columns held by each half-wave
#pragma unroll
    for (int r = 0; r < 8; r++) {
        ysum[r] += __shfl_xor(ysum[r], 1);
        ysum[r] += __shfl_xor(ysum[r], 2);
        ysum[r] += __shfl_xor(ysum[r], 4);
        ysum[r] += __shfl_xor(ysum[r], 8);
    }
    if (lcol == 0) {
        const float b2 = bp2[0];
        const int mofs = (lane < 16) ? 0 : 8;   // lanes<16 -> rows 0..7, lanes>=16 -> rows 8..15
#pragma unroll
        for (int r = 0; r < 8; r++) y[tile * 16 + mofs + r] = ysum[r] + b2;
    }
}

// ---------------------------------------------------------------------------
// Host-side orchestration
// ---------------------------------------------------------------------------
namespace {
constexpr int NUc = 100000, NMc = 20000, Ec = 500000;
constexpr size_t OFF_XL    = 0;
constexpr size_t OFF_XR    = OFF_XL    + (size_t)NUc * 128;
constexpr size_t OFF_ACC   = OFF_XR    + (size_t)NUc * 128;
constexpr size_t OFF_UCOMB = OFF_ACC   + (size_t)NUc * 128;
constexpr size_t OFF_MCOMB = OFF_UCOMB + (size_t)NUc * 128;
constexpr size_t OFF_UFIN  = OFF_MCOMB + (size_t)NMc * 128;
constexpr size_t OFF_MFIN  = OFF_UFIN  + (size_t)NUc * 64;
constexpr size_t OFF_LOG   = OFF_MFIN  + (size_t)NMc * 64;
constexpr size_t OFF_EX    = OFF_LOG   + (size_t)Ec * 2;
constexpr size_t OFF_MENC  = OFF_EX    + (size_t)Ec * 2;
constexpr size_t OFF_DEN   = OFF_MENC  + (size_t)NUc * 2;
constexpr size_t OFF_WT    = OFF_DEN   + (size_t)NUc * 2;   // 128*128 bf16 = 8192 floats

inline void run_conv(const float* xs, int Ns, const float* xd, int Nd,
                     const int* gl, const int* gs,
                     const float* Wl, const float* bl, const float* Wr, const float* br,
                     const float* att, const float* bias,
                     float* comb, int colofs, float* aux,
                     float* w, hipStream_t stream) {
    float*          xl      = w + OFF_XL;
    float*          xr      = w + OFF_XR;
    float*          acc     = w + OFF_ACC;
    float*          logitsb = w + OFF_LOG;
    float*          exb     = w + OFF_EX;
    unsigned*       m_enc   = (unsigned*)(w + OFF_MENC);
    float*          den     = w + OFF_DEN;
    unsigned short* wt      = (unsigned short*)(w + OFF_WT);

    pack_wt_kernel<<<(128 * 128 + 255) / 256, 256, 0, stream>>>(Wl, wt, 128);
    wmma_gemm_kernel<<<(Ns / 16 + 7) / 8, 256, 0, stream>>>(xs, wt, bl, xl, Ns, 128, 0);
    pack_wt_kernel<<<(128 * 128 + 255) / 256, 256, 0, stream>>>(Wr, wt, 128);
    wmma_gemm_kernel<<<(Nd / 16 + 7) / 8, 256, 0, stream>>>(xd, wt, br, xr, Nd, 128, 0);
    conv_init_kernel<<<(Nd * 128 + 255) / 256, 256, 0, stream>>>(acc, m_enc, den, Nd);
    logits_kernel<<<(Ec * 2 + 255) / 256, 256, 0, stream>>>(xl, xr, gl, gs, att, logitsb, m_enc, Ec);
    mfin_kernel<<<(Nd * 2 + 255) / 256, 256, 0, stream>>>(m_enc, Nd);
    exden_kernel<<<(Ec * 2 + 255) / 256, 256, 0, stream>>>(logitsb, (float*)m_enc, gs, exb, den, Ec);
    scatter_kernel<<<(Ec * 2 + 255) / 256, 256, 0, stream>>>(xl, exb, den, gl, gs, acc, Ec);
    headmean_kernel<<<(Nd * 64 + 255) / 256, 256, 0, stream>>>(acc, bias, comb, colofs, aux, Nd);
}
} // namespace

extern "C" void kernel_launch(void* const* d_in, const int* in_sizes, int n_in,
                              void* d_out, int out_size, void* d_ws, size_t ws_size,
                              hipStream_t stream) {
    (void)in_sizes; (void)n_in; (void)out_size; (void)ws_size;

    const float* x_u = (const float*)d_in[0];
    const float* x_m = (const float*)d_in[1];
    const int*   src = (const int*)d_in[2];
    const int*   dst = (const int*)d_in[3];
    // per-conv weight packs: Wl, bl, Wr, br, att, bias
    const float* P[4][6];
    for (int cv = 0; cv < 4; cv++)
        for (int j = 0; j < 6; j++)
            P[cv][j] = (const float*)d_in[4 + cv * 6 + j];
    const float* Wuf = (const float*)d_in[28]; const float* buf_ = (const float*)d_in[29];
    const float* Wmf = (const float*)d_in[30]; const float* bmf = (const float*)d_in[31];
    const float* Wp1 = (const float*)d_in[32]; const float* bp1 = (const float*)d_in[33];
    const float* Wp2 = (const float*)d_in[34]; const float* bp2 = (const float*)d_in[35];

    float* w   = (float*)d_ws;
    float* out = (float*)d_out;
    float* y     = out;
    float* u_sat = out + Ec;
    float* u_dis = u_sat + (size_t)NUc * 64;
    float* m_sat = u_dis + (size_t)NUc * 64;
    float* m_dis = m_sat + (size_t)NMc * 64;

    float*          u_comb  = w + OFF_UCOMB;
    float*          m_comb  = w + OFF_MCOMB;
    float*          u_final = w + OFF_UFIN;
    float*          m_final = w + OFF_MFIN;
    unsigned short* wt      = (unsigned short*)(w + OFF_WT);

    // conv 0: um_sat (u->m)   conv 1: mu_sat (m->u, flipped edges)
    run_conv(x_u, NUc, x_m, NMc, src, dst, P[0][0], P[0][1], P[0][2], P[0][3], P[0][4], P[0][5],
             m_comb, 0, m_sat, w, stream);
    run_conv(x_m, NMc, x_u, NUc, dst, src, P[1][0], P[1][1], P[1][2], P[1][3], P[1][4], P[1][5],
             u_comb, 0, u_sat, w, stream);
    // conv 2: um_dis          conv 3: mu_dis
    run_conv(x_u, NUc, x_m, NMc, src, dst, P[2][0], P[2][1], P[2][2], P[2][3], P[2][4], P[2][5],
             m_comb, 64, m_dis, w, stream);
    run_conv(x_m, NMc, x_u, NUc, dst, src, P[3][0], P[3][1], P[3][2], P[3][3], P[3][4], P[3][5],
             u_comb, 64, u_dis, w, stream);

    // fusion GEMMs (ReLU)
    pack_wt_kernel<<<(128 * 64 + 255) / 256, 256, 0, stream>>>(Wuf, wt, 64);
    wmma_gemm_kernel<<<(NUc / 16 + 7) / 8, 256, 0, stream>>>(u_comb, wt, buf_, u_final, NUc, 64, 1);
    pack_wt_kernel<<<(128 * 64 + 255) / 256, 256, 0, stream>>>(Wmf, wt, 64);
    wmma_gemm_kernel<<<(NMc / 16 + 7) / 8, 256, 0, stream>>>(m_comb, wt, bmf, m_final, NMc, 64, 1);

    // edge predictor
    pack_wt_kernel<<<(128 * 64 + 255) / 256, 256, 0, stream>>>(Wp1, wt, 64);
    edge_mlp_kernel<<<(Ec / 16 + 7) / 8, 256, 0, stream>>>(u_final, m_final, src, dst,
                                                           wt, bp1, Wp2, bp2, y, Ec);
}
